// SpatialAttention_WH_4037269258661
// MI455X (gfx1250) — matrix operand order repaired and checked by reading, not compiled, not run on hardware
//
#include <hip/hip_runtime.h>
#include <hip/hip_bf16.h>
#include <math.h>

typedef __bf16 bf16;
typedef __attribute__((ext_vector_type(16))) __bf16 v16bf;
typedef __attribute__((ext_vector_type(8)))  float  v8f;

#define WMMA_BF16(a, b, c) \
  __builtin_amdgcn_wmma_f32_16x16x32_bf16(false, (a), false, (b), (short)0, (c), false, false)

union U32BF2 { unsigned int u; __bf16 h[2]; };

// ---------------------------------------------------------------------------
// Fragment loaders (wave32, 16-bit data), contiguous K order shared by A and B
// A 16x32 (MxK), row-major source, ld elems/row. lane: m=lane&15, half=lane>>4
//   VGPR v holds K = half*16 + 2v  (+0,+1)  -> one b32 load
// ---------------------------------------------------------------------------
__device__ __forceinline__ v16bf load_a_rowmajor(const bf16* base, int ld, int lane) {
  v16bf f;
  int m = lane & 15, half = lane >> 4;
  const bf16* row = base + m * ld;
#pragma unroll
  for (int v = 0; v < 8; ++v) {
    int k0 = (half << 4) + (v << 1);
    U32BF2 cv; cv.u = *(const unsigned int*)(row + k0);
    f[2 * v] = cv.h[0]; f[2 * v + 1] = cv.h[1];
  }
  return f;
}

// B 32x16 (KxN), stored n-major (element (k,n) at base[n*ld + k]).
//   VGPR v holds K = half*16 + 2v (+0,+1), n = lane&15  -> one b32 load
__device__ __forceinline__ v16bf load_b_nmajor(const bf16* base, long ld, int lane) {
  v16bf f;
  int n = lane & 15, half = lane >> 4;
  const bf16* col = base + (long)n * ld;
#pragma unroll
  for (int v = 0; v < 8; ++v) {
    int k0 = (half << 4) + (v << 1);
    U32BF2 cv; cv.u = *(const unsigned int*)(col + k0);
    f[2 * v] = cv.h[0]; f[2 * v + 1] = cv.h[1];
  }
  return f;
}

// Dims
#define NB 4
#define NC 256
#define NIC 128
#define NH 192
#define NW 192
#define HP 96
#define NT 9216      // 96*96
#define NP 2304      // 48*48
#define HWPX 36864   // 192*192

// ---------------------------------------------------------------------------
// 1) maxpool(2) of x, transpose to [B*Nt, C] bf16 (channel-minor for B-frags)
// ---------------------------------------------------------------------------
__global__ void k_pool_xpt(const float* __restrict__ x, bf16* __restrict__ xpt) {
  long t = blockIdx.x * (long)blockDim.x + threadIdx.x;
  if (t >= (long)NB * NT * NC) return;
  int c = t & 255;
  long n = t >> 8;
  int b = (int)(n / NT), nl = (int)(n % NT);
  int i = nl / HP, j = nl % HP;
  const float* p = x + ((long)(b * NC + c) * NH + 2 * i) * NW + 2 * j;
  float m0 = fmaxf(p[0], p[1]);
  float m1 = fmaxf(p[NW], p[NW + 1]);
  xpt[n * NC + c] = (bf16)fmaxf(m0, m1);
}

// ---------------------------------------------------------------------------
// 2) weight repacks (fp32 -> bf16)
// ---------------------------------------------------------------------------
__global__ void k_repack_w1(const float* __restrict__ tw, const float* __restrict__ tb,
                            const float* __restrict__ gw, const float* __restrict__ gb,
                            const float* __restrict__ pw, const float* __restrict__ pb,
                            bf16* __restrict__ Wcat, float* __restrict__ bias) {
  int t = blockIdx.x * blockDim.x + threadIdx.x;
  if (t >= 384 * 256) return;
  int m = t >> 8, k = t & 255;
  const float* src  = (m < 128) ? tw : (m < 256 ? gw : pw);
  const float* bsrc = (m < 128) ? tb : (m < 256 ? gb : pb);
  int mr = m & 127;
  Wcat[t] = (bf16)src[mr * 256 + k];
  if (k == 0) bias[m] = bsrc[mr];
}

// Wk[oc][tap*128+ic] = W_w[oc][ic][tap]  (K-order: tap-major, ic-minor)
__global__ void k_repack_w3(const float* __restrict__ Ww, bf16* __restrict__ Wk) {
  int t = blockIdx.x * blockDim.x + threadIdx.x;
  if (t >= 256 * 1152) return;
  int oc = t / 1152, k = t % 1152;
  int tap = k >> 7, ic = k & 127;
  Wk[t] = (bf16)Ww[(oc * 128 + ic) * 9 + tap];
}

// ---------------------------------------------------------------------------
// 3) fused 3-head 1x1 conv: D[384, 36864] = Wcat @ xp^T  (bf16 WMMA, f32 acc)
//    grid(2304, 3), block 256 (8 waves); wave -> one 16x16 tile
// ---------------------------------------------------------------------------
__global__ void k_conv1x1(const bf16* __restrict__ Wcat, const float* __restrict__ bias,
                          const bf16* __restrict__ xpt, bf16* __restrict__ theta,
                          bf16* __restrict__ gfull, bf16* __restrict__ phifull) {
  int lane = threadIdx.x & 31, wave = threadIdx.x >> 5;
  int mtile = blockIdx.y * 8 + wave;   // 0..23
  int ntile = blockIdx.x;              // 0..2303
  v8f acc = {0.f, 0.f, 0.f, 0.f, 0.f, 0.f, 0.f, 0.f};
  const bf16* abase = Wcat + mtile * 16 * 256;
  const bf16* bbase = xpt + (long)ntile * 16 * 256;
#pragma unroll
  for (int kk = 0; kk < 8; ++kk) {
    v16bf a = load_a_rowmajor(abase + kk * 32, 256, lane);
    v16bf b = load_b_nmajor(bbase + kk * 32, 256, lane);
    acc = WMMA_BF16(a, b, acc);
  }
  int half = lane >> 4;
  int n0 = ntile * 16 + (lane & 15);
  int bb = n0 / NT, nl = n0 % NT;
#pragma unroll
  for (int i = 0; i < 8; ++i) {
    int m = mtile * 16 + i + 8 * half;
    float val = acc[i] + bias[m];
    int hc = m & 127;
    if (m < 128)      theta[((long)bb * NT + nl) * NIC + hc]        = (bf16)val; // [B,Nt,IC]
    else if (m < 256) gfull[((long)(bb * NIC + hc)) * NT + nl]      = (bf16)val; // [B,IC,96,96]
    else              phifull[((long)(bb * NIC + hc)) * NT + nl]    = (bf16)val; // [B,IC,96,96]
  }
}

// ---------------------------------------------------------------------------
// 4) maxpool(2) of g/phi; g -> [B,IC,Np], phi -> [B,Np,IC]
// ---------------------------------------------------------------------------
__global__ void k_pool2(const bf16* __restrict__ gfull, const bf16* __restrict__ phifull,
                        bf16* __restrict__ g, bf16* __restrict__ phi) {
  int t = blockIdx.x * blockDim.x + threadIdx.x;
  if (t >= NB * NIC * NP) return;
  int ic = t & 127;
  int r = t >> 7;
  int np = r % NP, b = r / NP;
  int i = np / 48, j = np % 48;
  long o = ((long)(b * NIC + ic) * HP + 2 * i) * HP + 2 * j;
  const bf16* pg = gfull + o;
  const bf16* pp = phifull + o;
  float mg = fmaxf(fmaxf((float)pg[0], (float)pg[1]), fmaxf((float)pg[HP], (float)pg[HP + 1]));
  float mp = fmaxf(fmaxf((float)pp[0], (float)pp[1]), fmaxf((float)pp[HP], (float)pp[HP + 1]));
  g[(long)(b * NIC + ic) * NP + np]  = (bf16)mg;
  phi[((long)b * NP + np) * NIC + ic] = (bf16)mp;
}

// ---------------------------------------------------------------------------
// 5) flash attention: per wave, 16 queries x all 2304 keys in 32-key chunks.
//    GEMM1 f=theta@phi^T (K=128), online softmax, P staged via LDS into
//    A-fragment layout, GEMM2 O += P@g (K=32, N=128). All bf16 WMMA, f32 acc.
//    grid(144, 4), block 128 (4 waves, 1 query-tile each)
// ---------------------------------------------------------------------------
__global__ void k_attn(const bf16* __restrict__ theta, const bf16* __restrict__ phi,
                       const bf16* __restrict__ g, bf16* __restrict__ y96) {
  __shared__ bf16 pbuf[4 * 16 * 32];
  int lane = threadIdx.x & 31, wave = threadIdx.x >> 5;
  int b = blockIdx.y;
  int qtile = blockIdx.x * 4 + wave;   // 0..575
  bf16* pb = pbuf + wave * 512;
  int half = lane >> 4, nlane = lane & 15;

  const bf16* thbase = theta + ((long)b * NT + qtile * 16) * NIC;
  v16bf aq[4];
#pragma unroll
  for (int kk = 0; kk < 4; ++kk) aq[kk] = load_a_rowmajor(thbase + kk * 32, NIC, lane);

  const bf16* phib = phi + (long)b * NP * NIC;
  const bf16* gb   = g + (long)b * NIC * NP;

  v8f O[8];
#pragma unroll
  for (int j = 0; j < 8; ++j) O[j] = (v8f){0.f, 0.f, 0.f, 0.f, 0.f, 0.f, 0.f, 0.f};
  float rowM[8], rowL[8];
#pragma unroll
  for (int v = 0; v < 8; ++v) { rowM[v] = -1e30f; rowL[v] = 0.f; }

  for (int kt = 0; kt < NP / 32; ++kt) {
    v8f s0 = {0.f, 0.f, 0.f, 0.f, 0.f, 0.f, 0.f, 0.f};
    v8f s1 = {0.f, 0.f, 0.f, 0.f, 0.f, 0.f, 0.f, 0.f};
#pragma unroll
    for (int kk = 0; kk < 4; ++kk) {
      v16bf b0 = load_b_nmajor(phib + (long)(kt * 32) * NIC + kk * 32, NIC, lane);
      v16bf b1 = load_b_nmajor(phib + (long)(kt * 32 + 16) * NIC + kk * 32, NIC, lane);
      s0 = WMMA_BF16(aq[kk], b0, s0);
      s1 = WMMA_BF16(aq[kk], b1, s1);
    }
    float scale[8];
#pragma unroll
    for (int v = 0; v < 8; ++v) {
      float mx = fmaxf(s0[v], s1[v]);
#pragma unroll
      for (int off = 1; off < 16; off <<= 1) mx = fmaxf(mx, __shfl_xor(mx, off, 32));
      float newM = fmaxf(rowM[v], mx);
      scale[v] = __expf(rowM[v] - newM);
      float p0 = __expf(s0[v] - newM);
      float p1 = __expf(s1[v] - newM);
      float rs = p0 + p1;
#pragma unroll
      for (int off = 1; off < 16; off <<= 1) rs += __shfl_xor(rs, off, 32);
      rowL[v] = rowL[v] * scale[v] + rs;
      rowM[v] = newM;
      int m = v + 8 * half;
      pb[m * 32 + nlane]      = (bf16)p0;   // P tile [16 x 32] row-major in LDS
      pb[m * 32 + 16 + nlane] = (bf16)p1;
    }
#pragma unroll
    for (int j = 0; j < 8; ++j)
#pragma unroll
      for (int i = 0; i < 8; ++i) O[j][i] *= scale[i];
    v16bf pf = load_a_rowmajor(pb, 32, lane);     // intra-wave LDS RAW, DS in-order
#pragma unroll
    for (int j = 0; j < 8; ++j) {
      v16bf bg = load_b_nmajor(gb + (long)(16 * j) * NP + kt * 32, NP, lane);
      O[j] = WMMA_BF16(pf, bg, O[j]);
    }
  }
  bf16* out = y96 + ((long)b * NT + qtile * 16) * NIC;  // [B,Nt,IC]
#pragma unroll
  for (int i = 0; i < 8; ++i) {
    float inv = 1.0f / rowL[i];
    int m = i + 8 * half;
#pragma unroll
    for (int j = 0; j < 8; ++j) out[m * NIC + 16 * j + nlane] = (bf16)(O[j][i] * inv);
  }
}

// ---------------------------------------------------------------------------
// 6) bilinear x2 upsample, align_corners=True; NHWC in/out (bf16)
// ---------------------------------------------------------------------------
__global__ void k_upsample(const bf16* __restrict__ y96, bf16* __restrict__ yup) {
  long t = blockIdx.x * (long)blockDim.x + threadIdx.x;
  if (t >= (long)NB * HWPX * NIC) return;
  int ic = t & 127;
  long r = t >> 7;
  int ox = (int)(r % NW); r /= NW;
  int oy = (int)(r % NH); int b = (int)(r / NH);
  const float sf = 95.0f / 191.0f;
  float ri = oy * sf, ci = ox * sf;
  int r0 = (int)ri, c0 = (int)ci;
  int r1 = (r0 + 1 > 95) ? 95 : r0 + 1;
  int c1 = (c0 + 1 > 95) ? 95 : c0 + 1;
  float fr = ri - r0, fc = ci - c0;
  const bf16* base = y96 + (long)b * NT * NIC + ic;
  float v00 = (float)base[(r0 * HP + c0) * NIC];
  float v01 = (float)base[(r0 * HP + c1) * NIC];
  float v10 = (float)base[(r1 * HP + c0) * NIC];
  float v11 = (float)base[(r1 * HP + c1) * NIC];
  float top = v00 * (1.f - fc) + v01 * fc;
  float bot = v10 * (1.f - fc) + v11 * fc;
  yup[t] = (bf16)(top * (1.f - fr) + bot * fr);
}

// ---------------------------------------------------------------------------
// 7) 3x3 conv as implicit GEMM (M=256 oc, K=1152, N=pixels) + bias + BN +
//    residual; writes both outputs. grid(2304, 2, 4), block 256 (8 waves)
// ---------------------------------------------------------------------------
__global__ void k_conv3(const bf16* __restrict__ Wk, const float* __restrict__ wb,
                        const bf16* __restrict__ yup,
                        const float* __restrict__ gamma, const float* __restrict__ beta,
                        const float* __restrict__ mean, const float* __restrict__ var,
                        const float* __restrict__ x, float* __restrict__ out) {
  int lane = threadIdx.x & 31, wave = threadIdx.x >> 5;
  int octile = blockIdx.y * 8 + wave;    // 0..15
  int b = blockIdx.z;
  int p = blockIdx.x * 16 + (lane & 15);
  int py = p / NW, px = p % NW;
  int half = lane >> 4;
  const bf16* ybase = yup + (long)b * HWPX * NIC;
  v8f acc = {0.f, 0.f, 0.f, 0.f, 0.f, 0.f, 0.f, 0.f};
#pragma unroll 4
  for (int kk = 0; kk < 36; ++kk) {
    int tap = kk >> 2;
    int dy = tap / 3 - 1, dx = tap % 3 - 1;
    int yy = py + dy, xx = px + dx;
    bool valid = (yy >= 0) && (yy < NH) && (xx >= 0) && (xx < NW);
    int icb = (kk & 3) * 32;
    const bf16* col = ybase + (long)(yy * NW + xx) * NIC + icb;
    v16bf bf;
#pragma unroll
    for (int v = 0; v < 8; ++v) {
      int k0 = (half << 4) + (v << 1);
      U32BF2 cv; cv.u = valid ? *(const unsigned int*)(col + k0) : 0u;
      bf[2 * v] = cv.h[0]; bf[2 * v + 1] = cv.h[1];
    }
    v16bf af = load_a_rowmajor(Wk + octile * 16 * 1152 + kk * 32, 1152, lane);
    acc = WMMA_BF16(af, bf, acc);
  }
  const long HWC = (long)NC * HWPX;
#pragma unroll
  for (int i = 0; i < 8; ++i) {
    int oc = octile * 16 + i + 8 * half;
    float inv = gamma[oc] * rsqrtf(var[oc] + 1e-5f);
    float wy = (acc[i] + wb[oc] - mean[oc]) * inv + beta[oc];
    long idx = (long)b * HWC + (long)oc * HWPX + p;
    out[idx] = wy + x[idx];                 // output 0: wy + x
    out[(long)NB * HWC + idx] = wy;         // output 1: wy
  }
}

// ---------------------------------------------------------------------------
extern "C" void kernel_launch(void* const* d_in, const int* in_sizes, int n_in,
                              void* d_out, int out_size, void* d_ws, size_t ws_size,
                              hipStream_t stream) {
  const float* x       = (const float*)d_in[0];
  const float* g_w     = (const float*)d_in[1];
  const float* g_b     = (const float*)d_in[2];
  const float* theta_w = (const float*)d_in[3];
  const float* theta_b = (const float*)d_in[4];
  const float* phi_w   = (const float*)d_in[5];
  const float* phi_b   = (const float*)d_in[6];
  const float* W_w     = (const float*)d_in[7];
  const float* W_b     = (const float*)d_in[8];
  const float* bn_g    = (const float*)d_in[9];
  const float* bn_be   = (const float*)d_in[10];
  const float* bn_m    = (const float*)d_in[11];
  const float* bn_v    = (const float*)d_in[12];
  float* out = (float*)d_out;

  char* ws = (char*)d_ws;
  size_t off = 0;
  auto take = [&](size_t bytes) -> void* {
    void* p = ws + off;
    off = (off + bytes + 255) & ~(size_t)255;
    return p;
  };
  bf16* xpt     = (bf16*)take((size_t)NB * NT * NC * 2);
  bf16* theta   = (bf16*)take((size_t)NB * NT * NIC * 2);
  bf16* gfull   = (bf16*)take((size_t)NB * NIC * NT * 2);
  bf16* phifull = (bf16*)take((size_t)NB * NIC * NT * 2);
  bf16* gp      = (bf16*)take((size_t)NB * NIC * NP * 2);
  bf16* php     = (bf16*)take((size_t)NB * NP * NIC * 2);
  bf16* y96     = (bf16*)take((size_t)NB * NT * NIC * 2);
  bf16* yup     = (bf16*)take((size_t)NB * HWPX * NIC * 2);
  bf16* Wcat    = (bf16*)take((size_t)384 * 256 * 2);
  float* bias   = (float*)take((size_t)384 * 4);
  bf16* Wk      = (bf16*)take((size_t)256 * 1152 * 2);

  k_repack_w1<<<384, 256, 0, stream>>>(theta_w, theta_b, g_w, g_b, phi_w, phi_b, Wcat, bias);
  k_repack_w3<<<1152, 256, 0, stream>>>(W_w, Wk);
  k_pool_xpt<<<(NB * NT * NC) / 256, 256, 0, stream>>>(x, xpt);
  k_conv1x1<<<dim3(NT * NB / 16, 3, 1), 256, 0, stream>>>(Wcat, bias, xpt, theta, gfull, phifull);
  k_pool2<<<(NB * NIC * NP) / 256, 256, 0, stream>>>(gfull, phifull, gp, php);
  k_attn<<<dim3(144, NB, 1), 128, 0, stream>>>(theta, php, gp, y96);
  k_upsample<<<(int)(((long)NB * HWPX * NIC) / 256), 256, 0, stream>>>(y96, yup);
  k_conv3<<<dim3(HWPX / 16, 2, NB), 256, 0, stream>>>(Wk, W_b, yup, bn_g, bn_be, bn_m, bn_v, x, out);
}
